// LPA_69260642615567
// MI455X (gfx1250) — matrix operand order repaired and checked by reading, not compiled, not run on hardware
//
#include <hip/hip_runtime.h>
#include <hip/hip_bf16.h>
#include <math.h>
#include <stdint.h>

typedef _Float16 half_t;
typedef _Float16 h16 __attribute__((ext_vector_type(16)));
typedef _Float16 h8  __attribute__((ext_vector_type(8)));
typedef float    f8  __attribute__((ext_vector_type(8)));
typedef float    f4  __attribute__((ext_vector_type(4)));

union H16u { h16 v; h8 h[2]; };

// ---------------------------------------------------------------------------
// CDNA5 async global->LDS copy (ASYNCcnt path) + waits
// ---------------------------------------------------------------------------
__device__ __forceinline__ void async_ld16(unsigned lds_off, const void* gaddr) {
  asm volatile("global_load_async_to_lds_b128 %0, %1, off"
               :: "v"(lds_off), "v"(gaddr) : "memory");
}
__device__ __forceinline__ void wait_async_le2() {
  asm volatile("s_wait_asynccnt 0x2" ::: "memory");
}
__device__ __forceinline__ void wait_async_le0() {
  asm volatile("s_wait_asynccnt 0x0" ::: "memory");
}
__device__ __forceinline__ unsigned lds_off_of(const void* p) {
  return (unsigned)(uintptr_t)p;   // low 32 bits of flat addr = LDS byte offset
}

// ---------------------------------------------------------------------------
// WMMA wrapper: D = A(16x32 f16) x B(32x16 f16) + C(16x16 f32)
// ---------------------------------------------------------------------------
__device__ __forceinline__ f8 wmma32(h16 a, h16 b, f8 c) {
  return __builtin_amdgcn_wmma_f32_16x16x32_f16(
      /*neg_a=*/false, a, /*neg_b=*/false, b,
      /*c_mod=*/(short)0, c, /*reuse_a=*/false, /*reuse_b=*/false);
}

// A fragment (16x32) from row-major f16 [M, ld].
// ISA layout: lanes 0-15 -> row M=lane, K = {0..7, 16..23}
//             lanes 16-31 -> row M=lane-16, K = {8..15, 24..31}
__device__ __forceinline__ h16 load_a16(const half_t* __restrict__ A, int ld,
                                        int m0, int k0, int lane) {
  int m  = m0 + (lane & 15);
  int kb = k0 + ((lane >> 4) << 3);
  const half_t* p = A + (long)m * ld + kb;
  H16u u;
  u.h[0] = *(const h8*)(p);
  u.h[1] = *(const h8*)(p + 16);
  return u.v;
}

// A fragment (16x32) converted on the fly from row-major f32 [M, ld].
__device__ __forceinline__ h16 load_a32(const float* __restrict__ A, int ld,
                                        int m0, int k0, int lane) {
  int m  = m0 + (lane & 15);
  int kb = k0 + ((lane >> 4) << 3);
  const float* p = A + (long)m * ld + kb;
  f4 x0 = *(const f4*)(p);
  f4 x1 = *(const f4*)(p + 4);
  f4 x2 = *(const f4*)(p + 16);
  f4 x3 = *(const f4*)(p + 20);
  h16 r;
#pragma unroll
  for (int i = 0; i < 4; ++i) {
    r[i]      = (half_t)x0[i];
    r[4 + i]  = (half_t)x1[i];
    r[8 + i]  = (half_t)x2[i];
    r[12 + i] = (half_t)x3[i];
  }
  return r;
}

// B fragment (32x16) from an LDS tile stored [64 rows x ldh halfs] (K contiguous
// per output column). ISA layout: lanes 0-15 -> col N=lane, K={0..7, 16..23};
// lanes 16-31 -> col N=lane-16, K={8..15, 24..31}.
__device__ __forceinline__ h16 load_b_lds(const half_t* tile, int ldh,
                                          int n0, int k0, int lane) {
  const half_t* p = tile + (n0 + (lane & 15)) * ldh + k0 + ((lane >> 4) << 3);
  H16u u;
  u.h[0] = *(const h8*)(p);
  u.h[1] = *(const h8*)(p + 16);
  return u.v;
}

// ---------------------------------------------------------------------------
// 1) MaxPool 2x2 over [B*L, 16, 16, 32] -> f16 att_in [8192, 2048]
// ---------------------------------------------------------------------------
__global__ void k_pool(const float* __restrict__ x, half_t* __restrict__ att_in) {
  int idx = blockIdx.x * 256 + threadIdx.x;           // 16,777,216 total
  int c  = idx & 31;
  int ow = (idx >> 5) & 7;
  int oh = (idx >> 8) & 7;
  int bl = idx >> 11;
  const float* p = x + (long)bl * 8192 + (oh * 2) * 512 + (ow * 2) * 32 + c;
  float m = fmaxf(fmaxf(p[0], p[32]), fmaxf(p[512], p[544]));
  att_in[idx] = (half_t)m;
}

// ---------------------------------------------------------------------------
// 2) Transpose + cast weights: f32 [K, N] -> f16 [N, K]
// ---------------------------------------------------------------------------
__global__ void k_wT(const float* __restrict__ Wsrc, half_t* __restrict__ WT,
                     int K, int N) {
  int idx = blockIdx.x * 256 + threadIdx.x;
  if (idx >= K * N) return;
  int n  = idx / K;
  int kk = idx - n * K;
  WT[idx] = (half_t)Wsrc[(long)kk * N + n];
}

// ---------------------------------------------------------------------------
// 3) Positional encoding [1024, 512] f32
// ---------------------------------------------------------------------------
__global__ void k_pos(float* __restrict__ pos) {
  int idx = blockIdx.x * 256 + threadIdx.x;           // 524,288 total
  int l = idx >> 9;
  int i = idx & 511;
  float lti = logf(1.0e4f) / 255.0f;                  // num_timescales-1 = 255
  int j = (i < 256) ? i : (i - 256);
  float t = (float)l * expf(-(float)j * lti);
  pos[idx] = (i < 256) ? sinf(t) : cosf(t);
}

// ---------------------------------------------------------------------------
// 4) res = att_in[8192,2048]f16 @ W_inT + b_in + pos -> f32 [8192,512]
//    Block = 4 waves x 16 M-rows sharing a double-buffered 64x32 B tile in LDS
//    streamed via global_load_async_to_lds_b128.
// ---------------------------------------------------------------------------
__global__ void k_gemm_res(const half_t* __restrict__ A, const half_t* __restrict__ BT,
                           const float* __restrict__ b_in, const float* __restrict__ pos,
                           float* __restrict__ res) {
  __shared__ half_t tileB[2][2560];                   // 64 rows x 40 halfs (32+8 pad)
  const int t    = threadIdx.x;
  const int warp = t >> 5;
  const int lane = t & 31;
  const int mblk = blockIdx.x >> 3;                   // 128 M blocks of 64 rows
  const int n0   = (blockIdx.x & 7) << 6;             // 8 N strips of 64
  const int m0   = (mblk << 6) + (warp << 4);

  const unsigned base = lds_off_of(&tileB[0][0]);
  const int r = t >> 2, s = t & 3;                    // seg r in [0,32), s in [0,4)
  const unsigned lb0 = (unsigned)(r * 80 + s * 16);
  const unsigned lb1 = (unsigned)((r + 32) * 80 + s * 16);
  const char* g0 = (const char*)(BT + (long)(n0 + r) * 2048) + s * 16;
  const char* g1 = (const char*)(BT + (long)(n0 + r + 32) * 2048) + s * 16;

  async_ld16(base + lb0, g0);                         // preload K-tile 0 -> buf 0
  async_ld16(base + lb1, g1);

  f8 acc[4] = {};
  for (int i = 0; i < 64; ++i) {                      // K = 2048, 32 per step
    if (i + 1 < 64) {
      unsigned nb = base + (((i + 1) & 1) ? 5120u : 0u);
      long koff = (long)(i + 1) << 6;                 // 32 halfs = 64 bytes
      async_ld16(nb + lb0, g0 + koff);
      async_ld16(nb + lb1, g1 + koff);
      wait_async_le2();                               // tile i landed (this wave)
    } else {
      wait_async_le0();
    }
    __syncthreads();                                  // tile i landed (all waves)
    if (i + 2 < 64)
      __builtin_prefetch(A + (long)(m0 + (lane & 15)) * 2048 + ((i + 2) << 5), 0, 1);
    h16 a = load_a16(A, 2048, m0, i << 5, lane);
    const half_t* bt = tileB[i & 1];
#pragma unroll
    for (int tt = 0; tt < 4; ++tt)
      acc[tt] = wmma32(a, load_b_lds(bt, 40, tt * 16, 0, lane), acc[tt]);
    __syncthreads();                                  // done reading buf before reuse
  }

  int nl    = lane & 15;
  int mbase = m0 + ((lane >> 4) << 3);
  int lbase = mbase & 1023;                           // l = m % 1024
#pragma unroll
  for (int tt = 0; tt < 4; ++tt) {
    int n = n0 + tt * 16 + nl;
    float bb = b_in[n];
#pragma unroll
    for (int rr = 0; rr < 8; ++rr)
      res[(long)(mbase + rr) * 512 + n] = acc[tt][rr] + bb + pos[(long)(lbase + rr) * 512 + n];
  }
}

// ---------------------------------------------------------------------------
// 5) LayerNorm(512) with zero-row mask -> f16 ln [8192, 512]
// ---------------------------------------------------------------------------
__global__ void k_ln(const float* __restrict__ res, const float* __restrict__ gamma,
                     const float* __restrict__ beta, half_t* __restrict__ ln) {
  int row  = blockIdx.x * 8 + (threadIdx.x >> 5);     // 8192 rows
  int lane = threadIdx.x & 31;
  const float* p = res + (long)row * 512 + lane * 16;
  f4 v[4];
#pragma unroll
  for (int i = 0; i < 4; ++i) v[i] = *(const f4*)(p + i * 4);
  float s = 0.f, ss = 0.f;
#pragma unroll
  for (int i = 0; i < 4; ++i)
#pragma unroll
    for (int j = 0; j < 4; ++j) { float e = v[i][j]; s += e; ss += e * e; }
  for (int off = 16; off; off >>= 1) {
    s  += __shfl_xor(s,  off, 32);
    ss += __shfl_xor(ss, off, 32);
  }
  float mean = s * (1.f / 512.f);
  float var  = ss * (1.f / 512.f) - mean * mean;
  float rstd = rsqrtf(var + 1e-6f);
  float zm   = (s != 0.f) ? 1.f : 0.f;
  const float* g = gamma + lane * 16;
  const float* b = beta  + lane * 16;
  H16u o;
#pragma unroll
  for (int i = 0; i < 4; ++i)
#pragma unroll
    for (int j = 0; j < 4; ++j)
      o.v[i * 4 + j] = (half_t)((g[i * 4 + j] * (v[i][j] - mean) * rstd + b[i * 4 + j]) * zm);
  half_t* dst = ln + (long)row * 512 + lane * 16;
  *(h8*)dst       = o.h[0];
  *(h8*)(dst + 8) = o.h[1];
}

// ---------------------------------------------------------------------------
// 6) Q/K/V projections: ln[8192,512]f16 @ W{q,k,v}T + bias.
//    q,k -> [b,h,l,64] f16 (q scaled by d^-0.5); v -> transposed [b,h,64,l].
//    Same async double-buffered LDS B-tile scheme.
// ---------------------------------------------------------------------------
__global__ void k_qkv(const half_t* __restrict__ ln,
                      const half_t* __restrict__ WqT, const half_t* __restrict__ WkT,
                      const half_t* __restrict__ WvT,
                      const float* __restrict__ bq, const float* __restrict__ bk,
                      const float* __restrict__ bv,
                      half_t* __restrict__ q, half_t* __restrict__ k,
                      half_t* __restrict__ vT) {
  __shared__ half_t tileB[2][2560];
  const int t    = threadIdx.x;
  const int warp = t >> 5;
  const int lane = t & 31;
  const int which = blockIdx.x >> 10;                 // 0=q 1=k 2=v (1024 blocks each)
  const int rem   = blockIdx.x & 1023;
  const int mblk  = rem >> 3;
  const int n0    = (rem & 7) << 6;
  const int m0    = (mblk << 6) + (warp << 4);
  const half_t* BT = (which == 0) ? WqT : (which == 1) ? WkT : WvT;

  const unsigned base = lds_off_of(&tileB[0][0]);
  const int r = t >> 2, s = t & 3;
  const unsigned lb0 = (unsigned)(r * 80 + s * 16);
  const unsigned lb1 = (unsigned)((r + 32) * 80 + s * 16);
  const char* g0 = (const char*)(BT + (long)(n0 + r) * 512) + s * 16;
  const char* g1 = (const char*)(BT + (long)(n0 + r + 32) * 512) + s * 16;

  async_ld16(base + lb0, g0);
  async_ld16(base + lb1, g1);

  f8 acc[4] = {};
  for (int i = 0; i < 16; ++i) {                      // K = 512
    if (i + 1 < 16) {
      unsigned nb = base + (((i + 1) & 1) ? 5120u : 0u);
      long koff = (long)(i + 1) << 6;
      async_ld16(nb + lb0, g0 + koff);
      async_ld16(nb + lb1, g1 + koff);
      wait_async_le2();
    } else {
      wait_async_le0();
    }
    __syncthreads();
    h16 a = load_a16(ln, 512, m0, i << 5, lane);
    const half_t* bt = tileB[i & 1];
#pragma unroll
    for (int tt = 0; tt < 4; ++tt)
      acc[tt] = wmma32(a, load_b_lds(bt, 40, tt * 16, 0, lane), acc[tt]);
    __syncthreads();
  }

  int nl    = lane & 15;
  int mbase = m0 + ((lane >> 4) << 3);
  int b  = m0 >> 10;
  int l0 = mbase & 1023;
  if (which == 2) {
#pragma unroll
    for (int tt = 0; tt < 4; ++tt) {
      int n = n0 + tt * 16 + nl;
      int h = n >> 6, j = n & 63;
      float bb = bv[n];
      h8 o;
#pragma unroll
      for (int rr = 0; rr < 8; ++rr) o[rr] = (half_t)(acc[tt][rr] + bb);
      *(h8*)(vT + ((long)(b * 8 + h) * 64 + j) * 1024 + l0) = o;
    }
  } else {
    half_t* dst = (which == 0) ? q : k;
    const float* bp = (which == 0) ? bq : bk;
    float scale = (which == 0) ? 0.125f : 1.0f;       // d^-0.5, d=64
#pragma unroll
    for (int tt = 0; tt < 4; ++tt) {
      int n = n0 + tt * 16 + nl;
      int h = n >> 6, j = n & 63;
      float bb = bp[n];
#pragma unroll
      for (int rr = 0; rr < 8; ++rr)
        dst[((long)(b * 8 + h) * 1024 + (l0 + rr)) * 64 + j] = (half_t)((acc[tt][rr] + bb) * scale);
    }
  }
}

// ---------------------------------------------------------------------------
// 7) logits = q @ k^T + bias -> f32 weights buffer (pre-softmax).
//    Full-K key tile (64 keys x 64 halfs) staged once in LDS via async copy.
// ---------------------------------------------------------------------------
__global__ void k_logits(const half_t* __restrict__ q, const half_t* __restrict__ k,
                         const float* __restrict__ bias, float* __restrict__ wts) {
  __shared__ half_t tileK[64 * 72];                   // 64 rows x 72 halfs (64+8 pad)
  const int t    = threadIdx.x;
  const int warp = t >> 5;
  const int lane = t & 31;
  const int bh   = blockIdx.x >> 8;                   // 64 (b,h) pairs
  const int rem  = blockIdx.x & 255;
  const int mblk = rem >> 4;                          // 16 M blocks of 64 rows
  const int n0   = (rem & 15) << 6;                   // 16 N strips of 64 keys
  const int m0   = (mblk << 6) + (warp << 4);
  const half_t* qh = q + (long)bh * 1024 * 64;
  const half_t* kh = k + (long)bh * 1024 * 64;

  const unsigned base = lds_off_of(&tileK[0]);
#pragma unroll
  for (int ss = 0; ss < 4; ++ss) {                    // 512 segs of 16B, 4/thread
    int seg = t + (ss << 7);
    int rr = seg >> 3, cc = seg & 7;
    async_ld16(base + (unsigned)(rr * 144 + cc * 16),
               (const char*)(kh + (long)(n0 + rr) * 64) + cc * 16);
  }
  wait_async_le0();
  __syncthreads();

  f8 acc[4] = {};
#pragma unroll
  for (int i = 0; i < 2; ++i) {                       // K = 64
    int k0 = i << 5;
    h16 a = load_a16(qh, 64, m0, k0, lane);
#pragma unroll
    for (int tt = 0; tt < 4; ++tt)
      acc[tt] = wmma32(a, load_b_lds(tileK, 72, tt * 16, k0, lane), acc[tt]);
  }

  int bb    = bh >> 3;                                // batch
  int nl    = lane & 15;
  int mbase = m0 + ((lane >> 4) << 3);
  float* out = wts + (long)bh * 1024 * 1024;
#pragma unroll
  for (int tt = 0; tt < 4; ++tt) {
    int n = n0 + tt * 16 + nl;
    float bi = bias[bb * 1024 + n];
#pragma unroll
    for (int rr = 0; rr < 8; ++rr)
      out[(long)(mbase + rr) * 1024 + n] = acc[tt][rr] + bi;
  }
}

// ---------------------------------------------------------------------------
// 8) Row softmax in-place over [65536, 1024]; one wave per row.
// ---------------------------------------------------------------------------
__global__ void k_softmax(float* __restrict__ wts) {
  int row  = blockIdx.x * 8 + (threadIdx.x >> 5);
  int lane = threadIdx.x & 31;
  float* p = wts + (long)row * 1024 + lane * 32;
  f4 v[8];
#pragma unroll
  for (int i = 0; i < 8; ++i) v[i] = *(const f4*)(p + i * 4);
  float mx = -3.402823466e38f;
#pragma unroll
  for (int i = 0; i < 8; ++i)
#pragma unroll
    for (int j = 0; j < 4; ++j) mx = fmaxf(mx, v[i][j]);
  for (int off = 16; off; off >>= 1) mx = fmaxf(mx, __shfl_xor(mx, off, 32));
  float s = 0.f;
#pragma unroll
  for (int i = 0; i < 8; ++i)
#pragma unroll
    for (int j = 0; j < 4; ++j) { float e = __expf(v[i][j] - mx); v[i][j] = e; s += e; }
  for (int off = 16; off; off >>= 1) s += __shfl_xor(s, off, 32);
  float inv = 1.f / s;
#pragma unroll
  for (int i = 0; i < 8; ++i) {
#pragma unroll
    for (int j = 0; j < 4; ++j) v[i][j] *= inv;
    *(f4*)(p + i * 4) = v[i];
  }
}

// ---------------------------------------------------------------------------
// 9) att = softmax(W)[1024,1024]f32 @ V + res -> d_out [8,1024,512].
//    A-fragments converted f32->f16 in-register; V tile (64x32) async in LDS.
// ---------------------------------------------------------------------------
__global__ void k_att(const float* __restrict__ wts, const half_t* __restrict__ vT,
                      const float* __restrict__ res, float* __restrict__ out) {
  __shared__ half_t tileB[2][2560];
  const int t    = threadIdx.x;
  const int warp = t >> 5;
  const int lane = t & 31;
  const int bh   = blockIdx.x >> 4;                   // 64 (b,h) pairs
  const int mblk = blockIdx.x & 15;
  const int m0   = (mblk << 6) + (warp << 4);
  const float*  W  = wts + (long)bh * 1024 * 1024;
  const half_t* vh = vT + (long)bh * 64 * 1024;

  const unsigned base = lds_off_of(&tileB[0][0]);
  const int r = t >> 2, s = t & 3;
  const unsigned lb0 = (unsigned)(r * 80 + s * 16);
  const unsigned lb1 = (unsigned)((r + 32) * 80 + s * 16);
  const char* g0 = (const char*)(vh + (long)r * 1024) + s * 16;
  const char* g1 = (const char*)(vh + (long)(r + 32) * 1024) + s * 16;

  async_ld16(base + lb0, g0);
  async_ld16(base + lb1, g1);

  f8 acc[4] = {};
  for (int i = 0; i < 32; ++i) {                      // K = 1024
    if (i + 1 < 32) {
      unsigned nb = base + (((i + 1) & 1) ? 5120u : 0u);
      long koff = (long)(i + 1) << 6;
      async_ld16(nb + lb0, g0 + koff);
      async_ld16(nb + lb1, g1 + koff);
      wait_async_le2();
    } else {
      wait_async_le0();
    }
    __syncthreads();
    if (i + 2 < 32)
      __builtin_prefetch(W + (long)(m0 + (lane & 15)) * 1024 + ((i + 2) << 5), 0, 1);
    h16 a = load_a32(W, 1024, m0, i << 5, lane);
    const half_t* bt = tileB[i & 1];
#pragma unroll
    for (int tt = 0; tt < 4; ++tt)
      acc[tt] = wmma32(a, load_b_lds(bt, 40, tt * 16, 0, lane), acc[tt]);
    __syncthreads();
  }

  int b = bh >> 3, h = bh & 7;
  int nl    = lane & 15;
  int mbase = m0 + ((lane >> 4) << 3);
#pragma unroll
  for (int tt = 0; tt < 4; ++tt) {
    int col = h * 64 + tt * 16 + nl;
#pragma unroll
    for (int rr = 0; rr < 8; ++rr) {
      long ri = (long)(b * 1024 + mbase + rr) * 512 + col;
      out[ri] = acc[tt][rr] + res[ri];
    }
  }
}

// ---------------------------------------------------------------------------
extern "C" void kernel_launch(void* const* d_in, const int* in_sizes, int n_in,
                              void* d_out, int out_size, void* d_ws, size_t ws_size,
                              hipStream_t stream) {
  const float* x     = (const float*)d_in[0];
  const float* bias  = (const float*)d_in[1];
  const float* W_in  = (const float*)d_in[2];
  const float* b_in  = (const float*)d_in[3];
  const float* gamma = (const float*)d_in[4];
  const float* beta  = (const float*)d_in[5];
  const float* Wq    = (const float*)d_in[6];
  const float* bq    = (const float*)d_in[7];
  const float* Wk    = (const float*)d_in[8];
  const float* bk    = (const float*)d_in[9];
  const float* Wv    = (const float*)d_in[10];
  const float* bv    = (const float*)d_in[11];

  float* out_att = (float*)d_out;                     // 8*1024*512
  float* wts     = (float*)d_out + 4194304;           // 8*8*1024*1024

  char* ws = (char*)d_ws;
  half_t* att_in = (half_t*)ws; ws += 33554432;       // [8192,2048] f16
  half_t* W_inT  = (half_t*)ws; ws += 2097152;        // [512,2048]  f16
  half_t* WqT    = (half_t*)ws; ws += 524288;         // [512,512]   f16
  half_t* WkT    = (half_t*)ws; ws += 524288;
  half_t* WvT    = (half_t*)ws; ws += 524288;
  float*  pos    = (float*)ws;  ws += 2097152;        // [1024,512]  f32
  float*  res    = (float*)ws;  ws += 16777216;       // [8192,512]  f32
  half_t* lnb    = (half_t*)ws; ws += 8388608;        // [8192,512]  f16
  half_t* q      = (half_t*)ws; ws += 8388608;        // [8,8,1024,64] f16
  half_t* k      = (half_t*)ws; ws += 8388608;        // [8,8,1024,64] f16
  half_t* vT     = (half_t*)ws; ws += 8388608;        // [8,8,64,1024] f16

  k_pool<<<65536, 256, 0, stream>>>(x, att_in);
  k_wT<<<4096, 256, 0, stream>>>(W_in, W_inT, 2048, 512);
  k_wT<<<1024, 256, 0, stream>>>(Wq, WqT, 512, 512);
  k_wT<<<1024, 256, 0, stream>>>(Wk, WkT, 512, 512);
  k_wT<<<1024, 256, 0, stream>>>(Wv, WvT, 512, 512);
  k_pos<<<2048, 256, 0, stream>>>(pos);
  k_gemm_res<<<1024, 128, 0, stream>>>(att_in, W_inT, b_in, pos, res);
  k_ln<<<1024, 256, 0, stream>>>(res, gamma, beta, lnb);
  k_qkv<<<3072, 128, 0, stream>>>(lnb, WqT, WkT, WvT, bq, bk, bv, q, k, vT);
  k_logits<<<16384, 128, 0, stream>>>(q, k, bias, wts);
  k_softmax<<<8192, 256, 0, stream>>>(wts);
  k_att<<<1024, 128, 0, stream>>>(wts, vT, res, out_att);
}
